// TextGINConv_40089224741280
// MI455X (gfx1250) — matrix operand order repaired and checked by reading, not compiled, not run on hardware
//
#include <hip/hip_runtime.h>

#define B_ 8
#define N_ 256
#define H_ 128
#define E_ 32

typedef __attribute__((ext_vector_type(16))) __bf16 v16bf;
typedef __attribute__((ext_vector_type(8)))  float  v8f;
typedef __attribute__((ext_vector_type(4)))  unsigned int v4u;
typedef __attribute__((ext_vector_type(8)))  int    v8i;
typedef __attribute__((ext_vector_type(4)))  int    v4i;

union BF16x16 { v16bf v; unsigned int u[8]; };

// Tensor Data Mover: 5-arg builtin on ROCm 7.2, 6-arg on amdgpu-toolchain
// (therock headers ship amd_gfx1250_TDM.h only in the 6-arg toolchain).
#if __has_include(<hip/amd_detail/amd_gfx1250_TDM.h>)
#define TDM_LOAD(g0, g1, g2, g3) \
  __builtin_amdgcn_tensor_load_to_lds((g0), (g1), (g2), (g3), (v8i){0,0,0,0,0,0,0,0}, 0)
#else
#define TDM_LOAD(g0, g1, g2, g3) \
  __builtin_amdgcn_tensor_load_to_lds((g0), (g1), (g2), (g3), 0)
#endif

// f32 pair -> packed bf16 DWORD (round-half-up via +0x8000, merge with v_perm_b32)
__device__ __forceinline__ unsigned int pk2(float lo, float hi) {
  unsigned int a = __float_as_uint(lo) + 0x8000u;
  unsigned int b = __float_as_uint(hi) + 0x8000u;
  return __builtin_amdgcn_perm(b, a, 0x07060302u);  // {b.hi16, a.hi16}
}

// Build the 8 packed DWORDs of a 16-bit A fragment from 16 consecutive f32:
// lanes 0-15: V0-7 <- K[0..15]; lanes 16-31 (kb=8): V0-7 <- K[16..31].
__device__ __forceinline__ void a_frag_from_row(const float* row, int kb, BF16x16& aw) {
  float4 a0 = *(const float4*)(row + 2 * kb);
  float4 a1 = *(const float4*)(row + 2 * kb + 4);
  float4 a2 = *(const float4*)(row + 2 * kb + 8);
  float4 a3 = *(const float4*)(row + 2 * kb + 12);
  aw.u[0] = pk2(a0.x, a0.y); aw.u[1] = pk2(a0.z, a0.w);
  aw.u[2] = pk2(a1.x, a1.y); aw.u[3] = pk2(a1.z, a1.w);
  aw.u[4] = pk2(a2.x, a2.y); aw.u[5] = pk2(a2.z, a2.w);
  aw.u[6] = pk2(a3.x, a3.y); aw.u[7] = pk2(a3.z, a3.w);
}

// -------- Kernel 1: fused edge projection + message + aggregation ----------
// One workgroup per (b,i); TDM pulls e[b,i,:,:] (256x32 f32 = 32 KB) into LDS
// once; 8 waves each own a 16-wide H slice and run 16 j-chunk WMMAs.
__global__ __launch_bounds__(256)
void gin_edge_msg_kernel(const float* __restrict__ x,
                         const int*   __restrict__ adj,
                         const float* __restrict__ e,
                         const float* __restrict__ We,
                         const float* __restrict__ be,
                         float*       __restrict__ hbuf) {
  __shared__ float se[N_ * E_];          // 32 KB
  const int bi   = blockIdx.x;           // b*N + i
  const int b    = bi >> 8;
  const int i    = bi & (N_ - 1);
  const int wave = threadIdx.x >> 5;
  const int lane = threadIdx.x & 31;
  const int col  = lane & 15;
  const int half = lane >> 4;
  const int hcol = wave * 16 + col;
  const int kb   = half * 8;

  const size_t ebase   = (size_t)bi * N_ * E_;
  const int    adjbase = bi * N_;
  const size_t xbase   = (size_t)b * N_ * H_;

  // One TDM per workgroup: D# per CDNA5 ISA §8.3/8.4.
  if (wave == 0) {
    unsigned long long ga = (unsigned long long)(const void*)(e + ebase);
    unsigned la = (unsigned)(unsigned long long)(const void*)se;
    v4u g0 = { 1u,                                   // count=1 (valid user D#)
               la,                                   // lds_addr
               (unsigned)ga,                         // global_addr[31:0]
               ((unsigned)(ga >> 32) & 0x01FFFFFFu) | (2u << 30) }; // addr[56:32]|type=2
    v8i g1 = { (int)(2u << 16),                      // data_size = 4 bytes
               (int)((unsigned)E_ << 16),            // tensor_dim0 = 32
               (int)((unsigned)N_ << 16),            // tensor_dim1 = 256
               (int)((unsigned)E_ << 16),            // tile_dim0 = 32
               (int)N_,                              // tile_dim1 = 256, tile_dim2 = 0
               (int)E_,                              // tensor_dim0_stride = 32
               (int)((unsigned)(N_ * E_) << 16),     // tensor_dim1_stride lo16 (unused, 2D)
               0 };
    v4i gz = { 0, 0, 0, 0 };
    TDM_LOAD(g0, g1, gz, gz);
    __builtin_amdgcn_s_wait_tensorcnt(0);
  }
  __syncthreads();

  // B fragment: We[32,128] column tile hcol (contiguous k order, same as the A fragment:
  // lanes 0-15 hold K=0..15, lanes 16-31 hold K=16..31, 2 K per VGPR).
  BF16x16 bw;
#pragma unroll
  for (int v = 0; v < 8; ++v) {
    const int k = half * 16 + 2 * v;
    bw.u[v] = pk2(We[k * H_ + hcol], We[(k + 1) * H_ + hcol]);
  }
  const float bev = be[hcol];

  float acc = 0.0f;

  for (int jc = 0; jc < 16; ++jc) {
    const int j0 = jc * 16;
    // A fragment from LDS: row M = col of the 16x32 tile.
    BF16x16 aw;
    a_frag_from_row(se + (j0 + col) * E_, kb, aw);

    v8f c = {};
    c = __builtin_amdgcn_wmma_f32_16x16x32_bf16(false, aw.v, false, bw.v,
                                                (short)0, c, false, false);

    // adj mask for rows j0..j0+15 via one load + wave32 ballot (bits 0-15).
    const int adjv = adj[adjbase + j0 + col];
    const unsigned amask = (unsigned)__ballot(adjv != 0);

    // C/D layout: lane covers rows M = half*8 + r, column N = col.
#pragma unroll
    for (int r = 0; r < 8; ++r) {
      const int j = j0 + half * 8 + r;
      float mv = c[r] + bev + x[xbase + (size_t)j * H_ + hcol];
      mv = fmaxf(mv, 0.0f);
      acc += ((amask >> (half * 8 + r)) & 1u) ? mv : 0.0f;
    }
  }

  // combine row-halves (rows 0-7 in lanes 0-15, rows 8-15 in lanes 16-31)
  acc += __shfl_xor(acc, 16, 32);
  if (half == 0) {
    hbuf[(size_t)bi * H_ + hcol] = acc + x[xbase + (size_t)i * H_ + hcol];
  }
}

// -------- Kernel 2: fused MLP  out = relu(h@W1+b1)@W2 + b2 ----------------
// 128 workgroups x 16 rows; stage-1 result staged in LDS for stage-2 A frags.
__global__ __launch_bounds__(256)
void gin_mlp_kernel(const float* __restrict__ hbuf,
                    const float* __restrict__ W1, const float* __restrict__ b1,
                    const float* __restrict__ W2, const float* __restrict__ b2,
                    float*       __restrict__ out) {
  __shared__ float lds_t1[16 * 256];
  const int g0   = blockIdx.x * 16;     // row base (0..2047)
  const int wave = threadIdx.x >> 5;
  const int lane = threadIdx.x & 31;
  const int col  = lane & 15;
  const int half = lane >> 4;
  const int kb   = half * 8;

  // ---- stage 1: t1 = relu(h @ W1 + b1)   (16 x 256) ----
#pragma unroll
  for (int t = 0; t < 2; ++t) {
    const int nbase = (wave * 2 + t) * 16;
    v8f acc = {};
#pragma unroll
    for (int kc = 0; kc < 4; ++kc) {
      const int k0 = kc * 32;
      BF16x16 aw, bw;
      a_frag_from_row(hbuf + (size_t)(g0 + col) * H_ + k0, kb, aw);
#pragma unroll
      for (int v = 0; v < 8; ++v) {
        const int k = k0 + half * 16 + 2 * v;
        bw.u[v] = pk2(W1[k * (2 * H_) + nbase + col],
                      W1[(k + 1) * (2 * H_) + nbase + col]);
      }
      acc = __builtin_amdgcn_wmma_f32_16x16x32_bf16(false, aw.v, false, bw.v,
                                                    (short)0, acc, false, false);
    }
    const float b1v = b1[nbase + col];
#pragma unroll
    for (int r = 0; r < 8; ++r) {
      lds_t1[(half * 8 + r) * 256 + nbase + col] = fmaxf(acc[r] + b1v, 0.0f);
    }
  }
  __syncthreads();

  // ---- stage 2: out = t1 @ W2 + b2   (16 x 128) ----
  {
    const int nbase = wave * 16;
    v8f acc = {};
#pragma unroll
    for (int kc = 0; kc < 8; ++kc) {
      const int k0 = kc * 32;
      BF16x16 aw, bw;
      a_frag_from_row(&lds_t1[col * 256 + k0], kb, aw);
#pragma unroll
      for (int v = 0; v < 8; ++v) {
        const int k = k0 + half * 16 + 2 * v;
        bw.u[v] = pk2(W2[k * H_ + nbase + col],
                      W2[(k + 1) * H_ + nbase + col]);
      }
      acc = __builtin_amdgcn_wmma_f32_16x16x32_bf16(false, aw.v, false, bw.v,
                                                    (short)0, acc, false, false);
    }
    const float b2v = b2[nbase + col];
#pragma unroll
    for (int r = 0; r < 8; ++r) {
      out[(size_t)(g0 + half * 8 + r) * H_ + nbase + col] = acc[r] + b2v;
    }
  }
}

extern "C" void kernel_launch(void* const* d_in, const int* in_sizes, int n_in,
                              void* d_out, int out_size, void* d_ws, size_t ws_size,
                              hipStream_t stream) {
  (void)in_sizes; (void)n_in; (void)out_size; (void)ws_size;
  const float* x   = (const float*)d_in[0];
  const int*   adj = (const int*)  d_in[1];
  const float* e   = (const float*)d_in[2];
  const float* We  = (const float*)d_in[3];
  const float* be  = (const float*)d_in[4];
  const float* W1  = (const float*)d_in[5];
  const float* b1  = (const float*)d_in[6];
  const float* W2  = (const float*)d_in[7];
  const float* b2  = (const float*)d_in[8];
  float* outp = (float*)d_out;
  float* hbuf = (float*)d_ws;          // B*N*H fp32 = 1 MB scratch

  gin_edge_msg_kernel<<<B_ * N_, 256, 0, stream>>>(x, adj, e, We, be, hbuf);
  gin_mlp_kernel<<<(B_ * N_) / 16, 256, 0, stream>>>(hbuf, W1, b1, W2, b2, outp);
}
